// RGTLayer_51264729645646
// MI455X (gfx1250) — compile-verified, matrix operand order checked by reading, not run on hardware
//
#include <hip/hip_runtime.h>

typedef __attribute__((ext_vector_type(16))) _Float16 v16h;
typedef __attribute__((ext_vector_type(16))) float    v16f;
typedef __attribute__((ext_vector_type(8)))  float    v8f;

#define NEG_SLOPE 0.01f
#define LN_EPS 1e-6f

// ---------- helpers ----------

// order-preserving float -> uint key (for atomic max on floats incl. negatives)
__device__ __forceinline__ unsigned fkey(float f) {
    unsigned u = __float_as_uint(f);
    return (u & 0x80000000u) ? ~u : (u | 0x80000000u);
}
__device__ __forceinline__ float funkey(unsigned k) {
    unsigned u = (k & 0x80000000u) ? (k & 0x7fffffffu) : ~k;
    return __uint_as_float(u);
}

__device__ __forceinline__ v8f wmma_f16(v16h a, v16h b, v8f c) {
    return __builtin_amdgcn_wmma_f32_16x16x32_f16(
        /*neg_a=*/false, a, /*neg_b=*/false, b,
        /*c_mod=*/(short)0, c, /*reuse_a=*/false, /*reuse_b=*/false);
}

// Pack f32 chunks (kb..kb+7, kb+16..kb+23) into an fp16 fragment lane slice.
// Native packed cvt via __builtin_convertvector (cheap VALU, unlike manual bf16 RNE).
__device__ __forceinline__ v16h packA(float4 c0, float4 c1, float4 c2, float4 c3) {
    v16f t;
    t[0]  = c0.x; t[1]  = c0.y; t[2]  = c0.z; t[3]  = c0.w;
    t[4]  = c1.x; t[5]  = c1.y; t[6]  = c1.z; t[7]  = c1.w;
    t[8]  = c2.x; t[9]  = c2.y; t[10] = c2.z; t[11] = c2.w;
    t[12] = c3.x; t[13] = c3.y; t[14] = c3.z; t[15] = c3.w;
    return __builtin_convertvector(t, v16h);
}

__device__ __forceinline__ float leaky(float v) { return v > 0.f ? v : NEG_SLOPE * v; }

// ---------- kernel 0: init node accumulators ----------

__global__ void k_init(float* __restrict__ agg, float* __restrict__ denom,
                       unsigned* __restrict__ maxKey, int N) {
    int i = blockIdx.x * 256 + threadIdx.x;
    if (i < N * 64) agg[i] = 0.f;
    if (i < N) { denom[i] = 0.f; maxKey[i] = 0u; /* key 0 == -inf sentinel */ }
}

// ---------- kernel 1: edge GEMMs (fp16 WMMA, fp32 accum) + att + atomic max ----------
// 1 wave per 16-edge tile, 8 waves / block. Weights staged once per block in LDS
// as pre-swizzled fp16 B-fragments. Per tile: 3 GEMMs 16x128x64 = 48 WMMAs.

__global__ void __launch_bounds__(256)
k_edge_gemm(const float* __restrict__ h, const float* __restrict__ eh,
            const float* __restrict__ qrh, const float* __restrict__ qeh,
            const int* __restrict__ src, const int* __restrict__ dst,
            const float* __restrict__ W_msg, const float* __restrict__ W_k,
            const float* __restrict__ W_q,
            float* __restrict__ msgBuf, float* __restrict__ attBuf,
            unsigned* __restrict__ maxKey, int E) {
    // 3 gemms * 4 ksteps * 4 ntiles * 32 lanes * 16 fp16 = 48KB
    __shared__ alignas(32) _Float16 ldsW[3 * 16 * 32 * 16];
    __shared__ float attLds[8 * 16];

    const int tid = threadIdx.x;
    const float* Ws[3] = { W_msg, W_k, W_q };

    // cooperative fill: fragment (g,kk,nt), lane, elem -> W[k][n]
    for (int i = tid; i < 3 * 16 * 32 * 16; i += 256) {
        int g    = i / (16 * 512);
        int rem  = i % (16 * 512);
        int kk   = rem / (4 * 512);
        int rem2 = rem % (4 * 512);
        int nt   = rem2 / 512;
        int li   = rem2 % 512;
        int lane = li / 16, el = li % 16;
        int n = nt * 16 + (lane & 15);
        int k = kk * 32 + ((lane >= 16) ? 8 : 0) + (el & 7) + ((el >= 8) ? 16 : 0);
        ldsW[i] = (_Float16)Ws[g][k * 64 + n];
    }

    const int wave = tid >> 5;
    const int lane = tid & 31;
    const int e0 = (blockIdx.x * 8 + wave) * 16;
    const bool active = (e0 < E);

    __syncthreads();

    const v8f vzero = { 0.f, 0.f, 0.f, 0.f, 0.f, 0.f, 0.f, 0.f };
    v8f accM[4], accK[4], accQ[4];
    const int hi8 = (lane >= 16) ? 8 : 0;
    const int col = lane & 15;

    if (active) {
#pragma unroll
        for (int i = 0; i < 4; ++i) { accM[i] = vzero; accK[i] = vzero; accQ[i] = vzero; }

        int er = e0 + (lane & 15);
        if (er >= E) er = E - 1;
        const int s = src[er];
        const float* hrow  = h   + (size_t)s  * 64;
        const float* ehrow = eh  + (size_t)er * 64;
        const float* qrrow = qrh + (size_t)er * 64;
        const float* qerow = qeh + (size_t)er * 64;

#pragma unroll
        for (int kk = 0; kk < 4; ++kk) {
            const int kb = kk * 32 + hi8;
            const float* seBase = (kk < 2) ? (hrow + kb)  : (ehrow + (kb - 64));
            const float* qBase  = (kk < 2) ? (qrrow + kb) : (qerow + (kb - 64));

            float4 s0 = *(const float4*)(seBase);
            float4 s1 = *(const float4*)(seBase + 4);
            float4 s2 = *(const float4*)(seBase + 16);
            float4 s3 = *(const float4*)(seBase + 20);
            v16h aSe = packA(s0, s1, s2, s3);

            float4 q0 = *(const float4*)(qBase);
            float4 q1 = *(const float4*)(qBase + 4);
            float4 q2 = *(const float4*)(qBase + 16);
            float4 q3 = *(const float4*)(qBase + 20);
            v16h aQ = packA(q0, q1, q2, q3);

#pragma unroll
            for (int nt = 0; nt < 4; ++nt) {
                const v16h bM = *(const v16h*)&ldsW[((0 * 16 + kk * 4 + nt) * 32 + lane) * 16];
                accM[nt] = wmma_f16(aSe, bM, accM[nt]);
                const v16h bK = *(const v16h*)&ldsW[((1 * 16 + kk * 4 + nt) * 32 + lane) * 16];
                accK[nt] = wmma_f16(aSe, bK, accK[nt]);
                const v16h bQ = *(const v16h*)&ldsW[((2 * 16 + kk * 4 + nt) * 32 + lane) * 16];
                accQ[nt] = wmma_f16(aQ, bQ, accQ[nt]);
            }
        }

        // att[row] = (q . k) / sqrt(d): reduce over N (4 tiles in-lane, 16 lanes cross-lane)
        float p[8];
#pragma unroll
        for (int r = 0; r < 8; ++r) {
            float v = 0.f;
#pragma unroll
            for (int nt = 0; nt < 4; ++nt) v += accQ[nt][r] * accK[nt][r];
            v += __shfl_xor(v, 1, 16);
            v += __shfl_xor(v, 2, 16);
            v += __shfl_xor(v, 4, 16);
            v += __shfl_xor(v, 8, 16);
            p[r] = v * 0.125f;   // 1/sqrt(64)
        }
        if (lane == 0 || lane == 16) {
            const int base = wave * 16 + ((lane == 16) ? 8 : 0);
#pragma unroll
            for (int r = 0; r < 8; ++r) attLds[base + r] = p[r];
        }
    }

    __syncthreads();

    if (active) {
        if (lane < 16) {
            const int e = e0 + lane;
            if (e < E) {
                const float a = attLds[wave * 16 + lane];
                attBuf[e] = a;
                atomicMax(&maxKey[dst[e]], fkey(a));
            }
        }
        // store leaky-relu(msg)
#pragma unroll
        for (int nt = 0; nt < 4; ++nt) {
#pragma unroll
            for (int r = 0; r < 8; ++r) {
                const int e = e0 + r + hi8;
                if (e < E) msgBuf[(size_t)e * 64 + nt * 16 + col] = leaky(accM[nt][r]);
            }
        }
    }
}

// ---------- kernel 2: e=exp(att-m), denom += e, agg += e*msg (unnormalized) ----------

__global__ void k_edge_agg(const float* __restrict__ attBuf, const float* __restrict__ msgBuf,
                           const int* __restrict__ dst, const unsigned* __restrict__ maxKey,
                           float* __restrict__ denom, float* __restrict__ agg, int E) {
    const size_t idx = (size_t)blockIdx.x * 256 + threadIdx.x;
    if (idx >= (size_t)E * 64) return;
    const int edge = (int)(idx >> 6);
    const int dim  = (int)(idx & 63);
    const int d = dst[edge];
    const float m = funkey(maxKey[d]);
    const float ev = __expf(attBuf[edge] - m);
    if (dim == 0) atomicAdd(&denom[d], ev);
    atomicAdd(&agg[(size_t)d * 64 + dim], ev * msgBuf[idx]);
}

// ---------- kernel 3: (agg/denom)@W_out (WMMA) + leaky + residual + LayerNorm + mask ----------

__global__ void __launch_bounds__(256)
k_node_out(const float* __restrict__ agg, const float* __restrict__ denom,
           const float* __restrict__ h, const float* __restrict__ W_out,
           const float* __restrict__ gamma, const float* __restrict__ beta,
           float* __restrict__ out, int N) {
    __shared__ alignas(32) _Float16 ldsW[2 * 4 * 32 * 16]; // 8KB
    __shared__ float hnLds[8][16][64];                      // 32KB
    __shared__ float muLds[8][16];
    __shared__ float rsLds[8][16];

    const int tid = threadIdx.x;
    for (int i = tid; i < 2 * 4 * 32 * 16; i += 256) {
        int kk   = i / (4 * 512);
        int rem  = i % (4 * 512);
        int nt   = rem / 512;
        int li   = rem % 512;
        int lane = li / 16, el = li % 16;
        int n = nt * 16 + (lane & 15);
        int k = kk * 32 + ((lane >= 16) ? 8 : 0) + (el & 7) + ((el >= 8) ? 16 : 0);
        ldsW[i] = (_Float16)W_out[k * 64 + n];
    }
    __syncthreads();

    const int wave = tid >> 5;
    const int lane = tid & 31;
    const int n0 = (blockIdx.x * 8 + wave) * 16;
    const bool active = (n0 < N);
    const int hi8 = (lane >= 16) ? 8 : 0;
    const int col = lane & 15;

    if (active) {
        const v8f vzero = { 0.f, 0.f, 0.f, 0.f, 0.f, 0.f, 0.f, 0.f };
        v8f acc[4];
#pragma unroll
        for (int i = 0; i < 4; ++i) acc[i] = vzero;

        int row = n0 + (lane & 15);
        const int rowc = row < N ? row : N - 1;
        const float den = denom[rowc];
        const float invd = den > 0.f ? 1.f / den : 0.f;
        const float* ar = agg + (size_t)rowc * 64;

#pragma unroll
        for (int kk = 0; kk < 2; ++kk) {
            const int kb = kk * 32 + hi8;
            float4 c0 = *(const float4*)(ar + kb);
            float4 c1 = *(const float4*)(ar + kb + 4);
            float4 c2 = *(const float4*)(ar + kb + 16);
            float4 c3 = *(const float4*)(ar + kb + 20);
            c0.x *= invd; c0.y *= invd; c0.z *= invd; c0.w *= invd;
            c1.x *= invd; c1.y *= invd; c1.z *= invd; c1.w *= invd;
            c2.x *= invd; c2.y *= invd; c2.z *= invd; c2.w *= invd;
            c3.x *= invd; c3.y *= invd; c3.z *= invd; c3.w *= invd;
            const v16h a = packA(c0, c1, c2, c3);
#pragma unroll
            for (int nt = 0; nt < 4; ++nt) {
                const v16h b = *(const v16h*)&ldsW[((kk * 4 + nt) * 32 + lane) * 16];
                acc[nt] = wmma_f16(a, b, acc[nt]);
            }
        }

        // hn = leaky(gemm) + h, staged to LDS for the row-wise LayerNorm
#pragma unroll
        for (int nt = 0; nt < 4; ++nt) {
#pragma unroll
            for (int r = 0; r < 8; ++r) {
                const int m = r + hi8;
                const int node = n0 + m;
                const int c = nt * 16 + col;
                const float hb = h[(size_t)(node < N ? node : N - 1) * 64 + c];
                hnLds[wave][m][c] = leaky(acc[nt][r]) + hb;
            }
        }
    }

    __syncthreads();

    if (active && lane < 16) {
        const int m = lane;
        float mu = 0.f;
        for (int c = 0; c < 64; ++c) mu += hnLds[wave][m][c];
        mu *= (1.f / 64.f);
        float var = 0.f;
        for (int c = 0; c < 64; ++c) {
            const float dlt = hnLds[wave][m][c] - mu;
            var += dlt * dlt;
        }
        var *= (1.f / 64.f);
        muLds[wave][m] = mu;
        rsLds[wave][m] = rsqrtf(var + LN_EPS);
    }

    __syncthreads();

    if (active) {
        for (int j = 0; j < 32; ++j) {
            const int lin = j * 32 + lane;
            const int m = lin >> 6;
            const int c = lin & 63;
            const int node = n0 + m;
            if (node < N) {
                const float den = denom[node];
                float o;
                if (den > 0.f) {
                    o = (hnLds[wave][m][c] - muLds[wave][m]) * rsLds[wave][m] * gamma[c] + beta[c];
                } else {
                    o = h[(size_t)node * 64 + c];  // zero in-degree: pass-through
                }
                out[(size_t)node * 64 + c] = o;
            }
        }
    }
}

// ---------- launch ----------

extern "C" void kernel_launch(void* const* d_in, const int* in_sizes, int n_in,
                              void* d_out, int out_size, void* d_ws, size_t ws_size,
                              hipStream_t stream) {
    const float* h     = (const float*)d_in[0];
    const float* eh    = (const float*)d_in[1];
    const float* qrh   = (const float*)d_in[2];
    const float* qeh   = (const float*)d_in[3];
    const int*   src   = (const int*)d_in[4];
    const int*   dst   = (const int*)d_in[5];
    const float* W_msg = (const float*)d_in[6];
    const float* W_q   = (const float*)d_in[7];
    const float* W_k   = (const float*)d_in[8];
    const float* W_out = (const float*)d_in[9];
    const float* gamma = (const float*)d_in[10];
    const float* beta  = (const float*)d_in[11];

    const int N = in_sizes[0] / 64;
    const int E = in_sizes[4];

    char* ws = (char*)d_ws;
    size_t off = 0;
    float*    msgBuf = (float*)(ws + off);    off += (size_t)E * 64 * sizeof(float); // 204.8 MB
    float*    attBuf = (float*)(ws + off);    off += (size_t)E * sizeof(float);
    unsigned* maxKey = (unsigned*)(ws + off); off += (size_t)N * sizeof(unsigned);
    float*    denom  = (float*)(ws + off);    off += (size_t)N * sizeof(float);
    float*    agg    = (float*)(ws + off);

    // 0) init node accumulators
    {
        const int total = N * 64;
        k_init<<<(total + 255) / 256, 256, 0, stream>>>(agg, denom, maxKey, N);
    }
    // 1) edge GEMMs + att + atomic max
    {
        const int tiles  = (E + 15) / 16;
        const int blocks = (tiles + 7) / 8;
        k_edge_gemm<<<blocks, 256, 0, stream>>>(h, eh, qrh, qeh, src, dst,
                                                W_msg, W_k, W_q,
                                                msgBuf, attBuf, maxKey, E);
    }
    // 2) softmax numerator + unnormalized aggregation
    {
        const size_t total = (size_t)E * 64;
        const unsigned blocks = (unsigned)((total + 255) / 256);
        k_edge_agg<<<blocks, 256, 0, stream>>>(attBuf, msgBuf, dst, maxKey, denom, agg, E);
    }
    // 3) node output GEMM + LayerNorm + mask
    {
        const int ntiles = (N + 15) / 16;
        const int blocks = (ntiles + 7) / 8;
        k_node_out<<<blocks, 256, 0, stream>>>(agg, denom, h, W_out, gamma, beta,
                                               (float*)d_out, N);
    }
}
